// GNNLSTM_14783277432974
// MI455X (gfx1250) — matrix operand order repaired and checked by reading, not compiled, not run on hardware
//
#include <hip/hip_runtime.h>
#include <hip/hip_bf16.h>
#include <math.h>

// Problem constants (fixed by reference)
#define S_LEN  512      // sequence length (= N_NODES)
#define NBATCH 64       // LSTM batch
#define HID    128      // hidden / feature dim
#define GATES  512      // 4*HID
#define NN2    262144   // N_NODES^2 (fc output width)
#define WPITCH 136      // padded LDS row pitch in ushorts (128 data + 8 pad = 272B, 16B-aligned)
#define HPITCH 136      // same padded pitch for LDS h buffer (bank-conflict-free)

typedef __attribute__((ext_vector_type(2)))  float        v2f;
typedef __attribute__((ext_vector_type(4)))  float        v4f;
typedef __attribute__((ext_vector_type(8)))  float        v8f;
typedef __attribute__((ext_vector_type(16))) __bf16       v16bf;
typedef __attribute__((ext_vector_type(4)))  unsigned int v4u;
typedef __attribute__((ext_vector_type(4)))  int          v4i;
typedef __attribute__((ext_vector_type(8)))  int          v8i;

union BF16Frag { v16bf v; unsigned short h[16]; unsigned int u[8]; v4u q[2]; };

__device__ __forceinline__ unsigned short f2bf(float f) {
  unsigned int u = __float_as_uint(f);
  u += 0x7FFFu + ((u >> 16) & 1u);          // round-to-nearest-even
  return (unsigned short)(u >> 16);
}
__device__ __forceinline__ float sigmoidf_(float x) { return 1.0f / (1.0f + __expf(-x)); }
__device__ __forceinline__ v8f splat8(float s) {
  v8f r;
#pragma unroll
  for (int i = 0; i < 8; ++i) r[i] = s;
  return r;
}

// ---- Tensor Data Mover: DMA a [rows x 128] bf16 matrix from global into LDS,
// ---- inserting 4 pad DWORDs per 64-DWORD row (LDS pitch = 272B) during the copy.
// ---- D# layout per CDNA5 ISA.
__device__ __forceinline__ void tdm_load_w(const unsigned short* gsrc, unsigned int lds_byte_off,
                                           unsigned int rows, unsigned int cols) {
  unsigned long long ga = (unsigned long long)(uintptr_t)gsrc;
  v4u g0;
  g0[0] = 1u;                                              // count=1 (valid user descriptor)
  g0[1] = lds_byte_off;                                    // lds_addr
  g0[2] = (unsigned int)(ga & 0xFFFFFFFFu);                // global_addr[31:0]
  g0[3] = (unsigned int)((ga >> 32) & 0x01FFFFFFu) | (2u << 30);  // global_addr[56:32] | type=2
  v8i g1;
  unsigned int d0 = (1u << 16)      // data_size = 1 -> 2 bytes (bf16)
                  | (1u << 20)      // pad_enable
                  | (5u << 22)      // pad_interval code 5 -> every 64 DWORDs
                  | (3u << 25);     // pad_amount  code 3 -> 4 DWORDs (16B)
  g1[0] = (int)d0;
  g1[1] = (int)(cols << 16);                 // tensor_dim0[15:0] at bits 63:48
  g1[2] = (int)(rows << 16);                 // tensor_dim0 hi = 0 | tensor_dim1[15:0]
  g1[3] = (int)(cols << 16);                 // tensor_dim1 hi = 0 | tile_dim0
  g1[4] = (int)rows;                         // tile_dim1 | tile_dim2=0
  g1[5] = (int)cols;                         // tensor_dim0_stride[31:0]
  g1[6] = 0;                                 // stride hi | tensor_dim1_stride lo
  g1[7] = 0;
  v4i z4 = {0, 0, 0, 0};
#if defined(__clang_major__) && (__clang_major__ >= 23)
  v8i z8 = {0, 0, 0, 0, 0, 0, 0, 0};
  __builtin_amdgcn_tensor_load_to_lds(g0, g1, z4, z4, z8, 0);
#else
  __builtin_amdgcn_tensor_load_to_lds(g0, g1, z4, z4, 0);
#endif
}

// ---------------- GCN: degree / norm / scatter ----------------

__global__ void k_deg_init(float* deg, int T) {
  int i = blockIdx.x * blockDim.x + threadIdx.x;
  if (i < T) deg[i] = 1.0f;                  // self-loop weight
}

__global__ void k_deg_edges(const int* __restrict__ dst, const float* __restrict__ w,
                            float* deg, int E) {
  int e = blockIdx.x * blockDim.x + threadIdx.x;
  if (e < E) atomicAdd(&deg[dst[e]], w[e]);
}

__global__ void k_dinv(const float* __restrict__ deg, float* dinv, int T) {
  int i = blockIdx.x * blockDim.x + threadIdx.x;
  if (i < T) { float d = deg[i]; dinv[i] = (d > 0.0f) ? rsqrtf(d) : 0.0f; }
}

__global__ void k_self_bias(const float* __restrict__ XW, const float* __restrict__ dinv,
                            const float* __restrict__ bg, float* gout, int n) {
  int i = blockIdx.x * blockDim.x + threadIdx.x;
  if (i < n) {
    int t = i >> 7, c = i & 127;
    float di = dinv[t];
    gout[i] = bg[c] + di * di * XW[i];
  }
}

__global__ void k_scatter(const int* __restrict__ src, const int* __restrict__ dst,
                          const float* __restrict__ w, const float* __restrict__ dinv,
                          const float* __restrict__ XW, float* gout, int E) {
  int idx = blockIdx.x * blockDim.x + threadIdx.x;
  int e = idx >> 5, lane = idx & 31;
  if (e < E) {
    int sI = src[e], dI = dst[e];
    float norm = dinv[sI] * w[e] * dinv[dI];
    const float4 vv = *((const float4*)(XW + (size_t)sI * HID) + lane);
    float* o = gout + (size_t)dI * HID + (size_t)lane * 4;
    atomicAdd(o + 0, norm * vv.x);
    atomicAdd(o + 1, norm * vv.y);
    atomicAdd(o + 2, norm * vv.z);
    atomicAdd(o + 3, norm * vv.w);
  }
}

// ---------------- fp32 WMMA GEMM: XW = x @ W_gcn  (M=T, N=128, K=128) ----------------

__global__ __launch_bounds__(256) void k_gemm_xw(const float* __restrict__ X,
                                                 const float* __restrict__ W,
                                                 float* __restrict__ XW) {
  const int lane = threadIdx.x & 31, lm = lane & 15, hi = lane >> 4;
  const int wid = blockIdx.x * 8 + (threadIdx.x >> 5);
  const int mt = wid >> 3, nt = wid & 7;
  const float* arow = X + (size_t)(mt * 16 + lm) * HID;
  const float* bcol = W + nt * 16 + lm;
  v8f acc = splat8(0.0f);
#pragma unroll
  for (int kk = 0; kk < 32; ++kk) {
    const int k0 = kk * 4 + hi * 2;
    v2f a, b;
    a.x = arow[k0];               a.y = arow[k0 + 1];
    b.x = bcol[(size_t)k0 * HID]; b.y = bcol[(size_t)(k0 + 1) * HID];
    acc = __builtin_amdgcn_wmma_f32_16x16x4_f32(false, a, false, b, (short)0, acc, false, false);
  }
#pragma unroll
  for (int v = 0; v < 8; ++v)
    XW[(size_t)(mt * 16 + v + 8 * hi) * HID + nt * 16 + lm] = acc[v];
}

// ---------------- weight f32 -> bf16 staging ----------------

__global__ void k_f32_to_bf16(const float* __restrict__ src, unsigned short* __restrict__ dst, int n) {
  int i = blockIdx.x * blockDim.x + threadIdx.x;
  if (i < n) dst[i] = f2bf(src[i]);
}

// ---------------- persistent LSTM layer (1 workgroup, 32 waves) ----------------
// Weights DMA'd once into LDS by the TDM (with padding for conflict-free pitch),
// then 512 serial steps of gates[64,512] = xt@wih^T + h@whh^T + bias.
// wave w: mt=w>>3 (batch tile), jt=w&7 (hidden tile); its 4 accumulators are the
// i,f,g,o gate columns of the SAME (batch,hidden) elements -> cell math is
// wave-local, c-state lives in VGPRs, h round-trips through LDS as bf16.

__global__ __launch_bounds__(1024) void k_lstm_layer(
    const float* __restrict__ xs_f32,          // layer0 input: gcn out [T,H], row (b*S+s); or null
    const unsigned short* __restrict__ xs_bf,  // layer1 input: bf16 hs [S,B,H]; or null
    const float* __restrict__ bih, const float* __restrict__ bhh,          // [512]
    const unsigned short* __restrict__ wih_bf,                             // [512,128] bf16 global
    const unsigned short* __restrict__ whh_bf,                             // [512,128] bf16 global
    unsigned short* __restrict__ hs_out,                                   // [S,B,H] bf16
    float* __restrict__ last_out)                                          // [B,H] f32 or null
{
  __shared__ unsigned short wih_lds[GATES * WPITCH];   // 136 KB (padded pitch 272B)
  __shared__ unsigned short whh_lds[GATES * WPITCH];   // 136 KB
  __shared__ unsigned short h_lds[NBATCH * HPITCH];    // 17 KB
  __shared__ float bias_lds[GATES];                    // 2 KB

  const int tid  = threadIdx.x;
  const int lane = tid & 31;
  const int wave = tid >> 5;
  const int lm = lane & 15;
  const int hi = lane >> 4;
  const int mt = wave >> 3;          // batch tile
  const int jt = wave & 7;           // hidden-column tile

  // One wave issues both weight DMAs; TENSORcnt==0 before the barrier ensures
  // the data is resident for everyone after __syncthreads().
  if (tid == 0) {
    tdm_load_w(wih_bf, (unsigned int)(uintptr_t)(void*)wih_lds, GATES, HID);
    tdm_load_w(whh_bf, (unsigned int)(uintptr_t)(void*)whh_lds, GATES, HID);
    __builtin_amdgcn_s_wait_tensorcnt(0);
  }
  if (tid < GATES) bias_lds[tid] = bih[tid] + bhh[tid];
  for (int i = tid; i < NBATCH * HPITCH; i += 1024) h_lds[i] = 0;
  __syncthreads();

  const int b_a  = mt * 16 + lm;     // A-fragment row (batch index)
  const int jcol = jt * 16 + lm;     // D/B column within each gate

  float cst[8];
#pragma unroll
  for (int v = 0; v < 8; ++v) cst[v] = 0.0f;

  for (int s = 0; s < S_LEN; ++s) {
    v8f acc[4];
#pragma unroll
    for (int g = 0; g < 4; ++g) acc[g] = splat8(bias_lds[g * HID + jcol]);

#pragma unroll
    for (int kt = 0; kt < 4; ++kt) {
      // A fragment (16x32 bf16): lane m=lm; u[0..3] <-> K=kt*32+hi*8+{0..7},
      // u[4..7] <-> +16. Both halves are 16 contiguous bytes -> two b128 loads.
      BF16Frag ax;
      if (xs_f32) {
        const float* xr = xs_f32 + ((size_t)b_a * S_LEN + s) * HID;
        const int kb0 = kt * 32 + hi * 8;
        v4f f0 = *(const v4f*)(xr + kb0);
        v4f f1 = *(const v4f*)(xr + kb0 + 4);
        v4f f2 = *(const v4f*)(xr + kb0 + 16);
        v4f f3 = *(const v4f*)(xr + kb0 + 20);
#pragma unroll
        for (int j = 0; j < 4; ++j) {
          ax.h[j]      = f2bf(f0[j]);
          ax.h[4 + j]  = f2bf(f1[j]);
          ax.h[8 + j]  = f2bf(f2[j]);
          ax.h[12 + j] = f2bf(f3[j]);
        }
      } else {
        const char* xr = (const char*)(xs_bf + ((size_t)s * NBATCH + b_a) * HID);
        ax.q[0] = *(const v4u*)(xr + kt * 64 + hi * 16);
        ax.q[1] = *(const v4u*)(xr + kt * 64 + 32 + hi * 16);
      }
      BF16Frag ah;      // h fragment from padded LDS
      {
        const char* hr = (const char*)h_lds + (size_t)b_a * (HPITCH * 2);
        ah.q[0] = *(const v4u*)(hr + kt * 64 + hi * 16);
        ah.q[1] = *(const v4u*)(hr + kt * 64 + 32 + hi * 16);
      }
#pragma unroll
      for (int g = 0; g < 4; ++g) {
        const int ncol = g * HID + jcol;    // weight row = gate column
        // B 32x16: lane n=lm; u[0..3] <-> K=kt*32+hi*8+{0..7}, u[4..7] <-> +16 (matches A).
        const char* wr0 = (const char*)wih_lds + (size_t)ncol * (WPITCH * 2) + kt * 64 + hi * 16;
        const char* wr1 = (const char*)whh_lds + (size_t)ncol * (WPITCH * 2) + kt * 64 + hi * 16;
        BF16Frag bx, bh;
        bx.q[0] = *(const v4u*)(wr0);  bx.q[1] = *(const v4u*)(wr0 + 32);
        bh.q[0] = *(const v4u*)(wr1);  bh.q[1] = *(const v4u*)(wr1 + 32);
        acc[g] = __builtin_amdgcn_wmma_f32_16x16x32_bf16(false, ax.v, false, bx.v,
                                                         (short)0, acc[g], false, false);
        acc[g] = __builtin_amdgcn_wmma_f32_16x16x32_bf16(false, ah.v, false, bh.v,
                                                         (short)0, acc[g], false, false);
      }
    }

    // elementwise LSTM cell (registers only; torch gate order i,f,g,o)
    float hval[8];
#pragma unroll
    for (int v = 0; v < 8; ++v) {
      float iv = sigmoidf_(acc[0][v]);
      float fv = sigmoidf_(acc[1][v]);
      float gv = tanhf(acc[2][v]);
      float ov = sigmoidf_(acc[3][v]);
      cst[v]  = fv * cst[v] + iv * gv;
      hval[v] = ov * tanhf(cst[v]);
    }

    __syncthreads();   // everyone done reading h_lds for this step
#pragma unroll
    for (int v = 0; v < 8; ++v) {
      int bb = mt * 16 + v + 8 * hi;       // D row -> batch index
      unsigned short hb = f2bf(hval[v]);
      h_lds[bb * HPITCH + jcol] = hb;
      hs_out[((size_t)s * NBATCH + bb) * HID + jcol] = hb;
      if (last_out && s == S_LEN - 1) last_out[bb * HID + jcol] = hval[v];
    }
    __syncthreads();   // new h visible
  }
}

// ---------------- fp32 WMMA GEMM: out[64,262144] = last @ fc_w^T + fc_b ----------------

__global__ __launch_bounds__(256) void k_gemm_fc(const float* __restrict__ A,      // [64,128]
                                                 const float* __restrict__ Wt,     // [262144,128]
                                                 const float* __restrict__ bias,   // [262144]
                                                 float* __restrict__ out) {        // [64,262144]
  const int lane = threadIdx.x & 31, lm = lane & 15, hi = lane >> 4;
  const int wid = blockIdx.x * 8 + (threadIdx.x >> 5);
  const int mt = wid >> 14;            // 0..3
  const int nt = wid & 16383;          // 0..16383
  const float* arow = A  + (size_t)(mt * 16 + lm) * HID;
  const float* brow = Wt + (size_t)(nt * 16 + lm) * HID;   // B[k][n] = fc_w[n][k]
  v8f acc = splat8(bias[nt * 16 + lm]);
#pragma unroll
  for (int kk = 0; kk < 32; ++kk) {
    const int k0 = kk * 4 + hi * 2;
    v2f a, b;
    a.x = arow[k0]; a.y = arow[k0 + 1];
    b.x = brow[k0]; b.y = brow[k0 + 1];
    acc = __builtin_amdgcn_wmma_f32_16x16x4_f32(false, a, false, b, (short)0, acc, false, false);
  }
#pragma unroll
  for (int v = 0; v < 8; ++v)
    out[(size_t)(mt * 16 + v + 8 * hi) * (size_t)NN2 + (size_t)nt * 16 + lm] = acc[v];
}

// ---------------- launcher ----------------

extern "C" void kernel_launch(void* const* d_in, const int* in_sizes, int n_in,
                              void* d_out, int out_size, void* d_ws, size_t ws_size,
                              hipStream_t stream) {
  const float* x   = (const float*)d_in[0];
  const int*   ei  = (const int*)  d_in[1];   // [2,E]
  const float* ew  = (const float*)d_in[2];
  const float* Wg  = (const float*)d_in[4];
  const float* bg  = (const float*)d_in[5];
  const float* wih = (const float*)d_in[6];   // [2,512,128]
  const float* whh = (const float*)d_in[7];
  const float* bih = (const float*)d_in[8];   // [2,512]
  const float* bhh = (const float*)d_in[9];
  const float* fcw = (const float*)d_in[10];  // [262144,128]
  const float* fcb = (const float*)d_in[11];

  const int E = in_sizes[2];
  const int T = in_sizes[3];                  // 32768

  char* ws = (char*)d_ws;
  size_t off = 0;
  auto wsalloc = [&](size_t bytes) -> void* {
    void* p = ws + off;
    off = (off + bytes + 255) & ~(size_t)255;
    return p;
  };
  float*          deg  = (float*)wsalloc((size_t)T * 4);
  float*          dinv = (float*)wsalloc((size_t)T * 4);
  float*          XW   = (float*)wsalloc((size_t)T * HID * 4);
  float*          gout = (float*)wsalloc((size_t)T * HID * 4);
  unsigned short* wbf  = (unsigned short*)wsalloc((size_t)4 * GATES * HID * 2);
  unsigned short* hs0  = (unsigned short*)wsalloc((size_t)S_LEN * NBATCH * HID * 2);
  unsigned short* hs1  = (unsigned short*)wsalloc((size_t)S_LEN * NBATCH * HID * 2);
  float*          last = (float*)wsalloc((size_t)NBATCH * HID * 4);
  (void)ws_size; (void)n_in; (void)out_size;

  const int* srcI = ei;
  const int* dstI = ei + E;

  // GCN
  k_deg_init <<<(T + 255) / 256, 256, 0, stream>>>(deg, T);
  k_deg_edges<<<(E + 255) / 256, 256, 0, stream>>>(dstI, ew, deg, E);
  k_dinv     <<<(T + 255) / 256, 256, 0, stream>>>(deg, dinv, T);
  k_gemm_xw  <<<(T / 16) * (HID / 16) / 8, 256, 0, stream>>>(x, Wg, XW);
  k_self_bias<<<(T * HID) / 256, 256, 0, stream>>>(XW, dinv, bg, gout, T * HID);
  k_scatter  <<<((size_t)E * 32 + 255) / 256, 256, 0, stream>>>(srcI, dstI, ew, dinv, XW, gout, E);

  // LSTM weights -> bf16 (TDM source)
  const int WN = GATES * HID;  // 65536
  k_f32_to_bf16<<<WN / 256, 256, 0, stream>>>(wih,      wbf + 0 * WN, WN);  // wih0
  k_f32_to_bf16<<<WN / 256, 256, 0, stream>>>(whh,      wbf + 1 * WN, WN);  // whh0
  k_f32_to_bf16<<<WN / 256, 256, 0, stream>>>(wih + WN, wbf + 2 * WN, WN);  // wih1
  k_f32_to_bf16<<<WN / 256, 256, 0, stream>>>(whh + WN, wbf + 3 * WN, WN);  // whh1

  // Persistent LSTM layers (single workgroup each, time loop inside, TDM-staged weights)
  k_lstm_layer<<<1, 1024, 0, stream>>>(gout, nullptr, bih, bhh,
                                       wbf + 0 * WN, wbf + 1 * WN, hs0, nullptr);
  k_lstm_layer<<<1, 1024, 0, stream>>>(nullptr, hs0, bih + GATES, bhh + GATES,
                                       wbf + 2 * WN, wbf + 3 * WN, hs1, last);

  // FC readout (memory-bound on fc_w, fp32 WMMA)
  k_gemm_fc<<<(4 * (NN2 / 16)) / 8, 256, 0, stream>>>(last, fcw, fcb, (float*)d_out);
}
